// GCNConvNet_35716948034096
// MI455X (gfx1250) — matrix operand order repaired and checked by reading, not compiled, not run on hardware
//
#include <hip/hip_runtime.h>
#include <hip/hip_bf16.h>

typedef __attribute__((ext_vector_type(16))) __bf16       v16bf;
typedef __attribute__((ext_vector_type(8)))  float        v8f;
typedef __attribute__((ext_vector_type(8)))  unsigned int v8u;

#define HID 128
#define LDSP 136  // padded LDS row stride (16-bit elems); 272B keeps 16B alignment

__device__ inline unsigned int pack_bf16x2(float a, float b) {
    unsigned short lo = __builtin_bit_cast(unsigned short, (__bf16)a);
    unsigned short hi = __builtin_bit_cast(unsigned short, (__bf16)b);
    return (unsigned int)lo | ((unsigned int)hi << 16);
}

__device__ inline void atomic_add_f32(float* p, float v) {
    __hip_atomic_fetch_add(p, v, __ATOMIC_RELAXED, __HIP_MEMORY_SCOPE_AGENT);
}

// ---------------- degree / normalization ----------------
__global__ void kInitDeg(int* degI, int n) {
    int i = blockIdx.x * blockDim.x + threadIdx.x;
    if (i < n) degI[i] = 1;  // self-loop
}

__global__ void kDeg(const int* __restrict__ dst, int* degI, int E) {
    int e = blockIdx.x * blockDim.x + threadIdx.x;
    if (e < E) atomicAdd(&degI[dst[e]], 1);
}

__global__ void kDinv(const int* __restrict__ degI, float* dinv, int n) {
    int i = blockIdx.x * blockDim.x + threadIdx.x;
    if (i < n) dinv[i] = rsqrtf((float)degI[i]);
}

__global__ void kNorm(const int* __restrict__ src, const int* __restrict__ dst,
                      const float* __restrict__ dinv, float* normE, int E) {
    int e = blockIdx.x * blockDim.x + threadIdx.x;
    if (e < E) normE[e] = dinv[src[e]] * dinv[dst[e]];
}

// ---------------- f32 -> bf16 conversion (optionally fused relu) ----------------
__global__ void kCvtBf16(const float4* __restrict__ in, uint2* __restrict__ out,
                         int n4, int doRelu) {
    int i = blockIdx.x * blockDim.x + threadIdx.x;
    if (i >= n4) return;
    float4 v = in[i];
    if (doRelu) {
        v.x = fmaxf(v.x, 0.f); v.y = fmaxf(v.y, 0.f);
        v.z = fmaxf(v.z, 0.f); v.w = fmaxf(v.w, 0.f);
    }
    uint2 o;
    o.x = pack_bf16x2(v.x, v.y);
    o.y = pack_bf16x2(v.z, v.w);
    out[i] = o;
}

// ---------------- init accumulator with bias ----------------
__global__ void kBiasInit(float4* __restrict__ accum, const float4* __restrict__ bias4, int n4) {
    int i = blockIdx.x * blockDim.x + threadIdx.x;
    if (i >= n4) return;
    accum[i] = bias4[i & 31];  // 128 floats = 32 float4 per row
}

// ---------------- GEMM: HW = Hbf @ Wbf  (bf16 WMMA, f32 accumulate) ----------------
// Persistent blocks: 8 waves each own one 16-col strip; grid-stride over 16-row tiles.
// Fragments are assembled as packed uint dwords (2 bf16 per VGPR) -> no scratch traffic.
__global__ __launch_bounds__(256) void kGemm(const __bf16* __restrict__ Hbf,
                                             const __bf16* __restrict__ Wbf,
                                             float* __restrict__ HW, int nTiles) {
    __shared__ unsigned short sW[128 * LDSP];
    __shared__ unsigned short sA[16 * LDSP];
    const int t    = threadIdx.x;
    const int wave = t >> 5;
    const int lane = t & 31;
    const int half = lane >> 4;
    const int l16  = lane & 15;
    const int n0   = wave * 16;

    // Stage W (128x128 bf16 = 2048 uint4) into LDS, row-major [k][n]
    const uint4* Wg = (const uint4*)Wbf;
    for (int i = t; i < 2048; i += 256) {
        int row = i >> 4, col8 = i & 15;
        *(uint4*)&sW[row * LDSP + col8 * 8] = Wg[i];
    }
    __syncthreads();

    // Build B fragments once (held in registers for the whole kernel).
    // B 32x16: VGPR v holds K={2v,2v+1}; lanes 0-15 cover K 0-15, lanes 16-31 K 16-31.
    v16bf bfrag[4];
#pragma unroll
    for (int kc = 0; kc < 4; ++kc) {
        int kbase = 32 * kc + 16 * half;
        v8u bu;
#pragma unroll
        for (int v = 0; v < 8; ++v) {
            unsigned int lo = sW[(kbase + 2 * v) * LDSP + n0 + l16];
            unsigned int hi = sW[(kbase + 2 * v + 1) * LDSP + n0 + l16];
            bu[v] = lo | (hi << 16);
        }
        bfrag[kc] = __builtin_bit_cast(v16bf, bu);
    }

    for (int tile = blockIdx.x; tile < nTiles; tile += gridDim.x) {
        __syncthreads();  // protect sA reuse across iterations
        // Stage A tile: 16 rows x 128 K contiguous = 256 uint4, one per thread
        const uint4* Ag = (const uint4*)(Hbf + (size_t)tile * 16 * HID);
        {
            int row = t >> 4, col8 = t & 15;
            *(uint4*)&sA[row * LDSP + col8 * 8] = Ag[t];
        }
        __syncthreads();

        // A fragments: lanes 0-15 hold K 0..15; lanes 16-31 K 16..31.
        // K-pairs are contiguous in the LDS row -> single b32 load per VGPR.
        v8f c = {};
#pragma unroll
        for (int kc = 0; kc < 4; ++kc) {
            v8u au;
#pragma unroll
            for (int v = 0; v < 8; ++v) {
                int k = 32 * kc + 16 * half + 2 * v;
                au[v] = *(const unsigned int*)&sA[l16 * LDSP + k];
            }
            v16bf afrag = __builtin_bit_cast(v16bf, au);
            c = __builtin_amdgcn_wmma_f32_16x16x32_bf16(false, afrag, false, bfrag[kc],
                                                        (short)0, c, false, false);
        }

        // D layout: VGPR r -> row (r + 8*half), lanes 0-15 cols n0..n0+15
        const size_t rowBase = (size_t)tile * 16;
#pragma unroll
        for (int r = 0; r < 8; ++r) {
            int row = r + 8 * half;
            HW[(rowBase + row) * HID + n0 + l16] = c[r];
        }
    }
}

// ---------------- edge scatter: accum[dst] += norm * HW[src] ----------------
// One wave per edge; lane i handles features [4i, 4i+4).
__global__ void kEdgeScatter(const float* __restrict__ HW, const float* __restrict__ normE,
                             const int* __restrict__ src, const int* __restrict__ dst,
                             float* __restrict__ accum, int E) {
    int gid  = blockIdx.x * blockDim.x + threadIdx.x;
    int e    = gid >> 5;
    int lane = threadIdx.x & 31;
    if (e >= E) return;
    int   s = src[e], d = dst[e];
    float w = normE[e];
    float4 v = ((const float4*)(HW + (size_t)s * HID))[lane];
    float* out = accum + (size_t)d * HID + lane * 4;
    atomic_add_f32(out + 0, w * v.x);
    atomic_add_f32(out + 1, w * v.y);
    atomic_add_f32(out + 2, w * v.z);
    atomic_add_f32(out + 3, w * v.w);
}

// ---------------- self-loop: accum[i] += dinv[i]^2 * HW[i] (no atomics needed) ----
__global__ void kSelfLoop(const float4* __restrict__ HW4, const float* __restrict__ dinv,
                          float4* __restrict__ accum4, int n4) {
    int i = blockIdx.x * blockDim.x + threadIdx.x;
    if (i >= n4) return;
    float di = dinv[i >> 5];
    float w  = di * di;
    float4 v = HW4[i];
    float4 a = accum4[i];
    a.x += w * v.x; a.y += w * v.y; a.z += w * v.z; a.w += w * v.w;
    accum4[i] = a;
}

// ---------------- MLP head: relu(h@lw1+lb1) @ lw2 + lb2 -> sigmoid ----------------
__global__ void kHead(const float* __restrict__ h, const float* __restrict__ lw1,
                      const float* __restrict__ lb1, const float* __restrict__ lw2,
                      const float* __restrict__ lb2, float* __restrict__ out, int n) {
    __shared__ float srow[8][HID];
    int wave = threadIdx.x >> 5;
    int lane = threadIdx.x & 31;
    int node = blockIdx.x * 8 + wave;
    bool active = node < n;
    if (active) {
        const float* hr = h + (size_t)node * HID;
        for (int k = lane; k < HID; k += 32) srow[wave][k] = hr[k];
    }
    __syncthreads();
    if (!active) return;
    float a0 = lb1[lane], a1 = lb1[lane + 32];
#pragma unroll 4
    for (int k = 0; k < HID; ++k) {
        float hv = srow[wave][k];
        a0 = fmaf(hv, lw1[k * 64 + lane], a0);
        a1 = fmaf(hv, lw1[k * 64 + lane + 32], a1);
    }
    float z0 = fmaxf(a0, 0.f), z1 = fmaxf(a1, 0.f);
    float p  = z0 * lw2[lane] + z1 * lw2[lane + 32];
    for (int o = 16; o > 0; o >>= 1) p += __shfl_down(p, o, 32);
    if (lane == 0) out[node] = 1.f / (1.f + __expf(-(p + lb2[0])));
}

// =====================================================================
extern "C" void kernel_launch(void* const* d_in, const int* in_sizes, int n_in,
                              void* d_out, int out_size, void* d_ws, size_t ws_size,
                              hipStream_t stream) {
    const int N = in_sizes[0] / HID;
    const int E = in_sizes[1] / 2;

    const float* x    = (const float*)d_in[0];
    const int*   ei   = (const int*)d_in[1];
    const int*   src  = ei;
    const int*   dstp = ei + E;
    const float* w[4] = {(const float*)d_in[3], (const float*)d_in[5],
                         (const float*)d_in[7], (const float*)d_in[9]};
    const float* b[4] = {(const float*)d_in[4], (const float*)d_in[6],
                         (const float*)d_in[8], (const float*)d_in[10]};
    const float* lw1 = (const float*)d_in[11];
    const float* lb1 = (const float*)d_in[12];
    const float* lw2 = (const float*)d_in[13];
    const float* lb2 = (const float*)d_in[14];

    // workspace carve-up (~135 MB)
    size_t off = 0;
    auto take = [&](size_t bytes) -> void* {
        void* r = (char*)d_ws + off;
        off = (off + bytes + 255) & ~(size_t)255;
        return r;
    };
    int*    degI  = (int*)   take((size_t)N * 4);
    float*  dinv  = (float*) take((size_t)N * 4);
    float*  normE = (float*) take((size_t)E * 4);
    __bf16* Hbf   = (__bf16*)take((size_t)N * HID * 2);
    __bf16* Wbf   = (__bf16*)take((size_t)4 * HID * HID * 2);
    float*  HW    = (float*) take((size_t)N * HID * 4);
    float*  accum = (float*) take((size_t)N * HID * 4);

    const int B256   = 256;
    const int n4     = N * (HID / 4);  // N*32 float4 groups
    const int nTiles = (N + 15) / 16;
    const int gemmGrid = nTiles < 1024 ? nTiles : 1024;

    kInitDeg<<<(N + 255) / B256, B256, 0, stream>>>(degI, N);
    kDeg<<<(E + 255) / B256, B256, 0, stream>>>(dstp, degI, E);
    kDinv<<<(N + 255) / B256, B256, 0, stream>>>(degI, dinv, N);
    kNorm<<<(E + 255) / B256, B256, 0, stream>>>(src, dstp, dinv, normE, E);

    // x -> bf16 H0 ; weights -> bf16
    kCvtBf16<<<(n4 + 255) / B256, B256, 0, stream>>>((const float4*)x, (uint2*)Hbf, n4, 0);
    for (int l = 0; l < 4; ++l)
        kCvtBf16<<<(4096 + 255) / B256, B256, 0, stream>>>(
            (const float4*)w[l], (uint2*)(Wbf + (size_t)l * HID * HID), 4096, 0);

    for (int l = 0; l < 4; ++l) {
        kBiasInit<<<(n4 + 255) / B256, B256, 0, stream>>>((float4*)accum, (const float4*)b[l], n4);
        kGemm<<<gemmGrid, 256, 0, stream>>>(Hbf, Wbf + (size_t)l * HID * HID, HW, nTiles);
        long long ework = (long long)E * 32;
        kEdgeScatter<<<(int)((ework + 255) / 256), 256, 0, stream>>>(HW, normE, src, dstp, accum, E);
        kSelfLoop<<<(n4 + 255) / B256, B256, 0, stream>>>((const float4*)HW, dinv, (float4*)accum, n4);
        if (l < 3)
            kCvtBf16<<<(n4 + 255) / B256, B256, 0, stream>>>((const float4*)accum, (uint2*)Hbf, n4, 1);
    }

    kHead<<<(N + 7) / 8, 256, 0, stream>>>(accum, lw1, lb1, lw2, lb2, (float*)d_out, N);
}
